// ScaledDotProductAttention_50869592654113
// MI455X (gfx1250) — compile-verified, matrix operand order checked by reading, not run on hardware
//
#include <hip/hip_runtime.h>

#define DEV __device__ __forceinline__

typedef __attribute__((ext_vector_type(16))) __bf16 v16bf;
typedef __attribute__((ext_vector_type(8)))  float  v8f;
typedef unsigned int v4u __attribute__((ext_vector_type(4)));
typedef int v8i __attribute__((ext_vector_type(8)));
typedef int v4i __attribute__((ext_vector_type(4)));

// ---------- bf16 helpers ----------
DEV __bf16 f2bf(float x) {
  unsigned u = __builtin_bit_cast(unsigned, x);
  u += 0x7FFFu + ((u >> 16) & 1u);          // round-to-nearest-even
  unsigned short s = (unsigned short)(u >> 16);
  return __builtin_bit_cast(__bf16, s);
}

struct alignas(8) bf4 { __bf16 v[4]; };

DEV bf4 load4(const float* p) {
  float4 f = *(const float4*)p;
  bf4 r; r.v[0] = f2bf(f.x); r.v[1] = f2bf(f.y); r.v[2] = f2bf(f.z); r.v[3] = f2bf(f.w);
  return r;
}
DEV bf4 load4(const __bf16* p) { return *(const bf4*)p; }

DEV void storeOut(float* p, float v)  { *p = v; }
DEV void storeOut(__bf16* p, float v) { *p = f2bf(v); }

// ---------- WMMA fragment loaders (CDNA5 ISA 7.12.2) ----------
DEV int kmap(int e, int hi) { return ((e >> 3) << 4) + hi * 8 + (e & 7); }

DEV v16bf fragRow(const __bf16* p, int ld, int l, int hi) {
  v16bf f;
#pragma unroll
  for (int e = 0; e < 16; ++e) f[e] = p[l * ld + kmap(e, hi)];
  return f;
}
DEV v16bf fragCol(const __bf16* p, int ld, int l, int hi) {
  v16bf f;
#pragma unroll
  for (int e = 0; e < 16; ++e) f[e] = p[kmap(e, hi) * ld + l];
  return f;
}

DEV v8f wmma_bf16(v16bf a, v16bf b, v8f c) {
  return __builtin_amdgcn_wmma_f32_16x16x32_bf16(false, a, false, b, (short)0, c, false, false);
}

// ---------- Tensor Data Mover: 64x64 bf16 tile, row stride 1024 elems,
// LDS rows padded to 72 elems via D# pad feature (32 DWORDs data + 4 DWORDs pad)
#if __has_builtin(__builtin_amdgcn_tensor_load_to_lds)
#define USE_TDM 1
typedef __attribute__((address_space(3))) void lds_void_t;
DEV unsigned ldsAddrOf(const void* p) {
  return (unsigned)(unsigned long long)(lds_void_t*)(void*)p;
}
DEV void tdm_load_64x64(const __bf16* gsrc, const void* ldst) {
  unsigned long long ga = (unsigned long long)gsrc;
  v4u g0 = { 1u,                                       // count=1, user mode
             ldsAddrOf(ldst),                          // lds_addr (bytes)
             (unsigned)(ga & 0xFFFFFFFFu),             // global_addr[31:0]
             (unsigned)((ga >> 32) & 0x1FFFFFFu) | (2u << 30) };  // addr[56:32], type=2
  v8i g1;
  g1[0] = (1 << 16)      // data_size = 2 bytes
        | (1 << 20)      // pad_enable
        | (4 << 22)      // pad_interval: 32 DWORDs (one 64-elem bf16 row)
        | (3 << 25);     // pad_amount: 4 DWORDs (8 bf16 -> 72-elem LDS rows)
  g1[1] = (int)((1024u & 0xFFFFu) << 16);  // tensor_dim0 = 1024 (lo16)
  g1[2] = (int)(64u << 16);                // tensor_dim0 hi=0 | tensor_dim1=64 (lo16)
  g1[3] = (int)(64u << 16);                // tensor_dim1 hi=0 | tile_dim0=64
  g1[4] = 64;                              // tile_dim1=64, tile_dim2=0
  g1[5] = 1024;                            // tensor_dim0_stride = 1024 (lo32)
  g1[6] = 0;
  g1[7] = 0;
  v4i z4 = { 0, 0, 0, 0 };
  v8i z8 = { 0, 0, 0, 0, 0, 0, 0, 0 };
  __builtin_amdgcn_tensor_load_to_lds(g0, g1, z4, z4, z8, 0);
}
#else
#define USE_TDM 0
#endif

// =====================================================================
// GEMM: Y[M,N] = A[M,K] @ W[N,K]^T + bias[N]
// 128x64 block tile, BK=32, 256 threads (8 waves), double-buffered LDS,
// global loads register-staged so they overlap the WMMAs.
// =====================================================================
constexpr int BM = 128, BN = 64, BK = 32, LDS_LD = BK + 8;

template <typename AT, typename OT>
__global__ __launch_bounds__(256)
void gemm_wt_bias(const AT* __restrict__ A, const float* __restrict__ W,
                  const float* __restrict__ bias, OT* __restrict__ Y,
                  int M, int N, int K) {
  __shared__ __bf16 As[2][BM][LDS_LD];
  __shared__ __bf16 Bs[2][BN][LDS_LD];
  const int tid = threadIdx.x;
  const int lane = tid & 31, wv = tid >> 5;
  const int l = lane & 15, hi = lane >> 4;
  const int wr = wv & 3, wc = wv >> 2;
  const int m0 = blockIdx.y * BM, n0 = blockIdx.x * BN;

  bf4 aReg[4], bReg[2];
  auto loadRegs = [&](int k0) {
#pragma unroll
    for (int i = 0; i < 4; ++i) {
      int idx = tid + i * 256;
      aReg[i] = load4(&A[(size_t)(m0 + (idx >> 3)) * K + k0 + (idx & 7) * 4]);
    }
#pragma unroll
    for (int i = 0; i < 2; ++i) {
      int idx = tid + i * 256;
      bReg[i] = load4(&W[(size_t)(n0 + (idx >> 3)) * K + k0 + (idx & 7) * 4]);
    }
  };
  auto storeRegs = [&](int buf) {
#pragma unroll
    for (int i = 0; i < 4; ++i) {
      int idx = tid + i * 256;
      *(bf4*)&As[buf][idx >> 3][(idx & 7) * 4] = aReg[i];
    }
#pragma unroll
    for (int i = 0; i < 2; ++i) {
      int idx = tid + i * 256;
      *(bf4*)&Bs[buf][idx >> 3][(idx & 7) * 4] = bReg[i];
    }
  };

  v8f acc[2][2] = {};
  const int ktot = K / BK;
  loadRegs(0);
  storeRegs(0);
  __syncthreads();
  for (int kt = 0; kt < ktot; ++kt) {
    const int cur = kt & 1;
    if (kt + 1 < ktot) loadRegs((kt + 1) * BK);   // overlap with compute below
    v16bf af[2], bfr[2];
#pragma unroll
    for (int i = 0; i < 2; ++i) af[i]  = fragRow(&As[cur][wr * 32 + i * 16][0], LDS_LD, l, hi);
#pragma unroll
    for (int j = 0; j < 2; ++j) bfr[j] = fragRow(&Bs[cur][wc * 32 + j * 16][0], LDS_LD, l, hi);
#pragma unroll
    for (int i = 0; i < 2; ++i)
#pragma unroll
      for (int j = 0; j < 2; ++j)
        acc[i][j] = wmma_bf16(af[i], bfr[j], acc[i][j]);
    if (kt + 1 < ktot) storeRegs(1 - cur);        // other buffer: no race with readers of cur
    __syncthreads();
  }
#pragma unroll
  for (int i = 0; i < 2; ++i)
#pragma unroll
    for (int j = 0; j < 2; ++j) {
      int n = n0 + wc * 32 + j * 16 + l;
      float bv = bias[n];
#pragma unroll
      for (int r = 0; r < 8; ++r) {
        int m = m0 + wr * 32 + i * 16 + r + 8 * hi;
        storeOut(&Y[(size_t)m * N + n], acc[i][j][r] + bv);
      }
    }
}

// =====================================================================
// Flash attention: one (b, h, 64-query tile) per block; 4 waves.
// K/V tiles double-buffered in LDS; Tensor Data Mover software-pipelined:
// wave 0 issues tile kt+1 while the block computes on tile kt, waiting
// TENSORcnt<=2 (in-order completion => current tile's 2 loads are done).
// Online softmax fused with scale*weights*mask; O += P@V in bf16 WMMA.
// =====================================================================
constexpr int Hh = 16, NQc = 1024, NKc = 1024, DMc = 1024, DKc = 64;

__global__ __launch_bounds__(128)
void attn_flash(const __bf16* __restrict__ Qb, const __bf16* __restrict__ Kb,
                const __bf16* __restrict__ Vb, const unsigned char* __restrict__ msk,
                const float* __restrict__ wts, __bf16* __restrict__ Ob) {
  __shared__ __bf16 Ks[2][64][72];
  __shared__ __bf16 Vs[2][64][72];
  __shared__ __bf16 Ps[64][72];
  const int tid = threadIdx.x;
  const int lane = tid & 31, wv = tid >> 5;
  const int l = lane & 15, hi = lane >> 4;
  const int qt = blockIdx.x, h = blockIdx.y, b = blockIdx.z;
  const int qBase = qt * 64;
  const size_t bhRow = (size_t)(b * Hh + h) * NQc;
  const int ktot = NKc / 64;

  v16bf aQ[2];
#pragma unroll
  for (int ks = 0; ks < 2; ++ks) {
    const __bf16* qp = &Qb[(size_t)(b * NQc + qBase + wv * 16) * DMc + h * DKc + ks * 32];
    aQ[ks] = fragRow(qp, DMc, l, hi);
  }

  float mrow[8], lrow[8];
  v8f oAcc[4] = {};
#pragma unroll
  for (int r = 0; r < 8; ++r) { mrow[r] = -1e30f; lrow[r] = 0.f; }

#if USE_TDM
  if (wv == 0) {   // prologue: DMA tile 0 into buffer 0
    tdm_load_64x64(&Kb[(size_t)(b * NKc) * DMc + h * DKc], &Ks[0][0][0]);
    tdm_load_64x64(&Vb[(size_t)(b * NKc) * DMc + h * DKc], &Vs[0][0][0]);
  }
#endif

  for (int kt = 0; kt < ktot; ++kt) {
    const int cur = kt & 1;
#if USE_TDM
    if (wv == 0) {
      if (kt + 1 < ktot) {   // issue next tile into other buffer, keep it in flight
        tdm_load_64x64(&Kb[(size_t)(b * NKc + (kt + 1) * 64) * DMc + h * DKc], &Ks[1 - cur][0][0]);
        tdm_load_64x64(&Vb[(size_t)(b * NKc + (kt + 1) * 64) * DMc + h * DKc], &Vs[1 - cur][0][0]);
        __builtin_amdgcn_s_wait_tensorcnt(2);   // oldest 2 (= current tile) complete
      } else {
        __builtin_amdgcn_s_wait_tensorcnt(0);
      }
    }
#else
    {
      const __bf16* kSrc = &Kb[(size_t)(b * NKc + kt * 64) * DMc + h * DKc];
      const __bf16* vSrc = &Vb[(size_t)(b * NKc + kt * 64) * DMc + h * DKc];
      uint4 kTmp[4], vTmp[4];
#pragma unroll
      for (int i = 0; i < 4; ++i) {
        int idx = tid + i * 128;
        int r = idx >> 3, c = (idx & 7) * 8;
        kTmp[i] = *(const uint4*)&kSrc[(size_t)r * DMc + c];
        vTmp[i] = *(const uint4*)&vSrc[(size_t)r * DMc + c];
      }
#pragma unroll
      for (int i = 0; i < 4; ++i) {
        int idx = tid + i * 128;
        int r = idx >> 3, c = (idx & 7) * 8;
        *(uint4*)&Ks[cur][r][c] = kTmp[i];
        *(uint4*)&Vs[cur][r][c] = vTmp[i];
      }
    }
#endif
    // prefetch next key-tile's weights/mask (the bandwidth-dominant stream)
    if (kt + 1 < ktot) {
      __builtin_prefetch(&wts[(bhRow + qBase + (tid >> 1)) * NKc + (kt + 1) * 64 + (tid & 1) * 32], 0, 1);
      if (tid < 64)
        __builtin_prefetch(&msk[(bhRow + qBase + tid) * NKc + (kt + 1) * 64], 0, 1);
    }
    __syncthreads();   // publish current buffer

    // S = Q @ K^T : 4 column tiles x 2 k-steps
    v8f sAcc[4] = {};
#pragma unroll
    for (int j = 0; j < 4; ++j)
#pragma unroll
      for (int ks = 0; ks < 2; ++ks)
        sAcc[j] = wmma_bf16(aQ[ks], fragRow(&Ks[cur][j * 16][ks * 32], 72, l, hi), sAcc[j]);

    // fused scale * weights, mask -> -1e30 sentinel; loads batched per tile
    float sc[4][8];
#pragma unroll
    for (int j = 0; j < 4; ++j) {
      const size_t rowBase = bhRow + qBase + wv * 16 + 8 * hi;
      const int kcol = kt * 64 + j * 16 + l;
      const float* wp = &wts[rowBase * NKc + kcol];
      const unsigned char* mp = &msk[rowBase * NKc + kcol];
      float wf[8]; unsigned mb[8];
#pragma unroll
      for (int r = 0; r < 8; ++r) wf[r] = wp[(size_t)r * NKc];
#pragma unroll
      for (int r = 0; r < 8; ++r) mb[r] = mp[(size_t)r * NKc];
#pragma unroll
      for (int r = 0; r < 8; ++r) {
        float v = sAcc[j][r] * 0.125f * wf[r];   // 1/sqrt(64)
        sc[j][r] = mb[r] ? -1e30f : v;
      }
    }

    // online softmax: row reductions across the 16-lane half via shuffles
#pragma unroll
    for (int r = 0; r < 8; ++r) {
      float rmax = fmaxf(fmaxf(sc[0][r], sc[1][r]), fmaxf(sc[2][r], sc[3][r]));
#pragma unroll
      for (int s = 1; s < 16; s <<= 1) rmax = fmaxf(rmax, __shfl_xor(rmax, s));
      float mnew = fmaxf(mrow[r], rmax);
      float alpha = __expf(mrow[r] - mnew);
      float psum = 0.f;
#pragma unroll
      for (int j = 0; j < 4; ++j) {
        float p = __expf(sc[j][r] - mnew);
        psum += p;
        Ps[wv * 16 + r + 8 * hi][j * 16 + l] = f2bf(p);   // wave-private transpose via LDS
      }
#pragma unroll
      for (int s = 1; s < 16; s <<= 1) psum += __shfl_xor(psum, s);
      lrow[r] = lrow[r] * alpha + psum;
      mrow[r] = mnew;
#pragma unroll
      for (int j = 0; j < 4; ++j) oAcc[j][r] *= alpha;
    }

    // O += P @ V
#pragma unroll
    for (int ks = 0; ks < 2; ++ks) {
      v16bf aP = fragRow(&Ps[wv * 16][ks * 32], 72, l, hi);
#pragma unroll
      for (int j = 0; j < 4; ++j)
        oAcc[j] = wmma_bf16(aP, fragCol(&Vs[cur][ks * 32][j * 16], 72, l, hi), oAcc[j]);
    }
    __syncthreads();   // all reads of buffer `cur` done before it is re-targeted
  }

#pragma unroll
  for (int j = 0; j < 4; ++j)
#pragma unroll
    for (int r = 0; r < 8; ++r) {
      int qrow = qBase + wv * 16 + r + 8 * hi;
      int col = h * DKc + j * 16 + l;
      Ob[(size_t)(b * NQc + qrow) * DMc + col] = f2bf(oAcc[j][r] / lrow[r]);
    }
}

// =====================================================================
extern "C" void kernel_launch(void* const* d_in, const int* in_sizes, int n_in,
                              void* d_out, int out_size, void* d_ws, size_t ws_size,
                              hipStream_t stream) {
  const float* queries = (const float*)d_in[0];
  const float* keys    = (const float*)d_in[1];
  const float* values  = (const float*)d_in[2];
  const unsigned char* mask = (const unsigned char*)d_in[3];
  const float* wts = (const float*)d_in[4];
  const float* Wq = (const float*)d_in[5];
  const float* bq = (const float*)d_in[6];
  const float* Wk = (const float*)d_in[7];
  const float* bk = (const float*)d_in[8];
  const float* Wv = (const float*)d_in[9];
  const float* bv = (const float*)d_in[10];
  const float* Wo = (const float*)d_in[11];
  const float* bo = (const float*)d_in[12];

  const int Mrows = 8 * 1024;               // B*NQ
  const int DM = 1024;
  const size_t SZ = (size_t)Mrows * DM;
  __bf16* Qb = (__bf16*)d_ws;
  __bf16* Kb = Qb + SZ;
  __bf16* Vb = Kb + SZ;
  __bf16* Ob = Vb + SZ;

  dim3 gGemm(DM / BN, Mrows / BM);
  gemm_wt_bias<float, __bf16><<<gGemm, 256, 0, stream>>>(queries, Wq, bq, Qb, Mrows, DM, DM);
  gemm_wt_bias<float, __bf16><<<gGemm, 256, 0, stream>>>(keys,    Wk, bk, Kb, Mrows, DM, DM);
  gemm_wt_bias<float, __bf16><<<gGemm, 256, 0, stream>>>(values,  Wv, bv, Vb, Mrows, DM, DM);

  dim3 gAttn(NQc / 64, Hh, 8);
  attn_flash<<<gAttn, 128, 0, stream>>>(Qb, Kb, Vb, mask, wts, Ob);

  gemm_wt_bias<__bf16, float><<<gGemm, 256, 0, stream>>>(Ob, Wo, bo, (float*)d_out, Mrows, DM, DM);
}
